// SentenceAttention_28269474742694
// MI455X (gfx1250) — compile-verified, matrix operand order checked by reading, not run on hardware
//
#include <hip/hip_runtime.h>
#include <hip/hip_bf16.h>

typedef __attribute__((ext_vector_type(16))) _Float16 v16h;
typedef __attribute__((ext_vector_type(8)))  _Float16 v8h;
typedef __attribute__((ext_vector_type(4)))  _Float16 v4h;
typedef __attribute__((ext_vector_type(8)))  float    v8f;

#define Bn 128
#define Tn 512
#define En 512
#define Hn 512
#define Cn 512
#define NCn 2

__device__ __forceinline__ int iclamp(int v, int lo, int hi) {
  return v < lo ? lo : (v > hi ? hi : v);
}

__device__ __forceinline__ float sigmoidf_(float x) {
  return 1.0f / (1.0f + __expf(-x));
}

__device__ __forceinline__ float seluf_(float x) {
  const float scale = 1.0507009873554804934193349852946f;
  const float alpha = 1.6732632423543772848170429916717f;
  return x > 0.0f ? scale * x : scale * alpha * (__expf(x) - 1.0f);
}

// 16x32 f16 WMMA fragment from f32 data.
// ISA layout: lanes 0-15 hold K={0..7,16..23}, lanes 16-31 hold K={8..15,24..31}.
__device__ __forceinline__ v16h frag_f32(const float* __restrict__ rowp, int lane) {
  const int ko = (lane & 16) ? 8 : 0;
  float4 f0 = *(const float4*)(rowp + ko);
  float4 f1 = *(const float4*)(rowp + ko + 4);
  float4 f2 = *(const float4*)(rowp + ko + 16);
  float4 f3 = *(const float4*)(rowp + ko + 20);
  v16h a;
  a[0]  = (_Float16)f0.x; a[1]  = (_Float16)f0.y; a[2]  = (_Float16)f0.z; a[3]  = (_Float16)f0.w;
  a[4]  = (_Float16)f1.x; a[5]  = (_Float16)f1.y; a[6]  = (_Float16)f1.z; a[7]  = (_Float16)f1.w;
  a[8]  = (_Float16)f2.x; a[9]  = (_Float16)f2.y; a[10] = (_Float16)f2.z; a[11] = (_Float16)f2.w;
  a[12] = (_Float16)f3.x; a[13] = (_Float16)f3.y; a[14] = (_Float16)f3.z; a[15] = (_Float16)f3.w;
  return a;
}

// Same fragment from f16 data (two 16B vector loads, no conversion).
__device__ __forceinline__ v16h frag_f16(const _Float16* __restrict__ rowp, int lane) {
  const int ko = (lane & 16) ? 8 : 0;
  v8h lo = *(const v8h*)(rowp + ko);
  v8h hi = *(const v8h*)(rowp + ko + 16);
  v16h a;
#pragma unroll
  for (int i = 0; i < 8; ++i) { a[i] = lo[i]; a[8 + i] = hi[i]; }
  return a;
}

__global__ __launch_bounds__(256) void init_kernel(float* __restrict__ hbuf, int n) {
  int i = blockIdx.x * blockDim.x + threadIdx.x;
  if (i < n) hbuf[i] = 0.0f;
}

// Vectorized f32 -> f16 conversion (n4 = n/4 float4 groups).
__global__ __launch_bounds__(256) void cvt_kernel(
    const float* __restrict__ src, _Float16* __restrict__ dst, int n4) {
  int i = blockIdx.x * blockDim.x + threadIdx.x;
  if (i < n4) {
    float4 f = ((const float4*)src)[i];
    v4h h;
    h[0] = (_Float16)f.x; h[1] = (_Float16)f.y;
    h[2] = (_Float16)f.z; h[3] = (_Float16)f.w;
    ((v4h*)dst)[i] = h;
  }
}

// One GRU time step, both directions (blockIdx.y), 32 hidden columns per block.
// All 4 gate accumulators live simultaneously: 0=r(joint), 1=z(joint), 2=i_n, 3=h_n.
__global__ __launch_bounds__(256) void gru_step_kernel(
    int t,
    const int* __restrict__ tokens, const int* __restrict__ lengths,
    const float* __restrict__ emb,
    const _Float16* __restrict__ Wih16_f, const _Float16* __restrict__ Whh16_f,
    const float* __restrict__ bih_f, const float* __restrict__ bhh_f,
    const _Float16* __restrict__ Wih16_b, const _Float16* __restrict__ Whh16_b,
    const float* __restrict__ bih_b, const float* __restrict__ bhh_b,
    float* __restrict__ hbuf, _Float16* __restrict__ out_f, _Float16* __restrict__ out_b)
{
  __shared__ float g_lds[4][Bn][32];   // 64 KB
  const int dir  = blockIdx.y;
  const int c0   = blockIdx.x * 32;
  const int tid  = threadIdx.x;
  const int lane = tid & 31;
  const int wave = tid >> 5;           // 8 waves = 8 M-tiles of 16 batch rows
  const int m0   = wave * 16;
  const int nl   = lane & 15;
  const int mhi  = (lane & 16) ? 8 : 0;

  const _Float16* Wih = dir ? Wih16_b : Wih16_f;
  const _Float16* Whh = dir ? Whh16_b : Whh16_f;
  const float* bih = dir ? bih_b : bih_f;
  const float* bhh = dir ? bhh_b : bhh_f;
  float* h_src = hbuf + (size_t)(dir * 2 + (t & 1)) * (Bn * Hn);
  float* h_dst = hbuf + (size_t)(dir * 2 + ((t + 1) & 1)) * (Bn * Hn);
  _Float16* outp = dir ? out_b : out_f;

  // Per-lane A rows (batch rows of this wave's M-tile)
  const int brow  = m0 + nl;
  const int len_r = lengths[brow];
  const int teff  = dir ? iclamp(len_r - 1 - t, 0, Tn - 1) : t;
  const float* xrow = emb + (size_t)tokens[brow * Tn + teff] * En;
  const float* hrow = h_src + (size_t)brow * Hn;

  // ---- bias init (straight-line, no branches) ----
  v8f acc[4][2];
#pragma unroll
  for (int nt = 0; nt < 2; ++nt) {
    const int c = c0 + nt * 16 + nl;
    const float br = bih[c] + bhh[c];
    const float bz = bih[Hn + c] + bhh[Hn + c];
    const float bi = bih[2 * Hn + c];
    const float bh = bhh[2 * Hn + c];
#pragma unroll
    for (int i = 0; i < 8; ++i) {
      acc[0][nt][i] = br;
      acc[1][nt][i] = bz;
      acc[2][nt][i] = bi;
      acc[3][nt][i] = bh;
    }
  }

  // ---- input phase: x_t contributes to gates r, z, i_n ----
  for (int ks = 0; ks < En / 32; ++ks) {
    v16h a = frag_f32(xrow + ks * 32, lane);
#pragma unroll
    for (int nt = 0; nt < 2; ++nt) {
      const size_t cb = (size_t)(c0 + nt * 16 + nl);
      v16h b0 = frag_f16(Wih + cb * En + ks * 32, lane);
      acc[0][nt] = __builtin_amdgcn_wmma_f32_16x16x32_f16(
          false, a, false, b0, (short)0, acc[0][nt], false, false);
      v16h b1 = frag_f16(Wih + ((size_t)Hn + cb) * En + ks * 32, lane);
      acc[1][nt] = __builtin_amdgcn_wmma_f32_16x16x32_f16(
          false, a, false, b1, (short)0, acc[1][nt], false, false);
      v16h b2 = frag_f16(Wih + ((size_t)2 * Hn + cb) * En + ks * 32, lane);
      acc[2][nt] = __builtin_amdgcn_wmma_f32_16x16x32_f16(
          false, a, false, b2, (short)0, acc[2][nt], false, false);
    }
  }

  // ---- hidden phase: h_t contributes to gates r, z, h_n ----
  for (int ks = 0; ks < Hn / 32; ++ks) {
    v16h a = frag_f32(hrow + ks * 32, lane);
#pragma unroll
    for (int nt = 0; nt < 2; ++nt) {
      const size_t cb = (size_t)(c0 + nt * 16 + nl);
      v16h b0 = frag_f16(Whh + cb * Hn + ks * 32, lane);
      acc[0][nt] = __builtin_amdgcn_wmma_f32_16x16x32_f16(
          false, a, false, b0, (short)0, acc[0][nt], false, false);
      v16h b1 = frag_f16(Whh + ((size_t)Hn + cb) * Hn + ks * 32, lane);
      acc[1][nt] = __builtin_amdgcn_wmma_f32_16x16x32_f16(
          false, a, false, b1, (short)0, acc[1][nt], false, false);
      v16h b3 = frag_f16(Whh + ((size_t)2 * Hn + cb) * Hn + ks * 32, lane);
      acc[3][nt] = __builtin_amdgcn_wmma_f32_16x16x32_f16(
          false, a, false, b3, (short)0, acc[3][nt], false, false);
    }
  }

  // C/D layout: VGPR r, lanes<16 -> M=r, lanes>=16 -> M=8+r; N = lane%16
#pragma unroll
  for (int g = 0; g < 4; ++g)
#pragma unroll
    for (int nt = 0; nt < 2; ++nt)
#pragma unroll
      for (int r = 0; r < 8; ++r)
        g_lds[g][m0 + mhi + r][nt * 16 + nl] = acc[g][nt][r];
  __syncthreads();

  // Pointwise GRU update for this block's 128x32 slab
  for (int idx = tid; idx < Bn * 32; idx += 256) {
    const int b = idx >> 5;
    const int cl = idx & 31;
    const int c = c0 + cl;
    const float r  = sigmoidf_(g_lds[0][b][cl]);
    const float z  = sigmoidf_(g_lds[1][b][cl]);
    const float nn = tanhf(g_lds[2][b][cl] + r * g_lds[3][b][cl]);
    const float hold = h_src[(size_t)b * Hn + c];
    const float hnew = (1.0f - z) * nn + z * hold;
    const float h = (t < lengths[b]) ? hnew : hold;  // freeze on padded steps
    h_dst[(size_t)b * Hn + c] = h;
    outp[((size_t)b * Tn + t) * Hn + c] = (_Float16)h;
  }
}

// proj = tanh(out @ Wp.T + bp); scores = proj @ ctx (fused, no proj materialization)
__global__ __launch_bounds__(256) void proj_score_kernel(
    const int* __restrict__ lengths,
    const _Float16* __restrict__ out_f, const _Float16* __restrict__ out_b,
    const _Float16* __restrict__ Wp16, const float* __restrict__ bp,
    const float* __restrict__ ctx, float* __restrict__ scores)
{
  __shared__ float srow[16];
  const int tid  = threadIdx.x;
  const int lane = tid & 31;
  const int wave = tid >> 5;     // 8 waves x 64 cols = 512 = C
  const int nl   = lane & 15;
  const int mhi  = (lane & 16) ? 8 : 0;
  const int i0 = blockIdx.x * 16;      // 16 (b,t) rows, never crosses b (T%16==0)
  const int b  = i0 / Tn;
  const int t0 = i0 % Tn;
  if (tid < 16) srow[tid] = 0.0f;
  __syncthreads();

  const int len  = lengths[b];
  const int trow = t0 + nl;
  const int trev = iclamp(len - 1 - trow, 0, Tn - 1);
  const _Float16* rowf = out_f + ((size_t)b * Tn + trow) * Hn;
  const _Float16* rowb = out_b + ((size_t)b * Tn + trev) * Hn;

  v8f acc[4];
#pragma unroll
  for (int nt = 0; nt < 4; ++nt) {
    const float bv = bp[wave * 64 + nt * 16 + nl];
#pragma unroll
    for (int i = 0; i < 8; ++i) acc[nt][i] = bv;
  }
  for (int ks = 0; ks < Hn / 32; ++ks) {          // K = 0..511  (forward half)
    v16h a = frag_f16(rowf + ks * 32, lane);
#pragma unroll
    for (int nt = 0; nt < 4; ++nt) {
      const int n = wave * 64 + nt * 16 + nl;
      v16h bmat = frag_f16(Wp16 + (size_t)n * (2 * Hn) + ks * 32, lane);
      acc[nt] = __builtin_amdgcn_wmma_f32_16x16x32_f16(
          false, a, false, bmat, (short)0, acc[nt], false, false);
    }
  }
  for (int ks = 0; ks < Hn / 32; ++ks) {          // K = 512..1023 (backward half)
    v16h a = frag_f16(rowb + ks * 32, lane);
#pragma unroll
    for (int nt = 0; nt < 4; ++nt) {
      const int n = wave * 64 + nt * 16 + nl;
      v16h bmat = frag_f16(Wp16 + (size_t)n * (2 * Hn) + Hn + ks * 32, lane);
      acc[nt] = __builtin_amdgcn_wmma_f32_16x16x32_f16(
          false, a, false, bmat, (short)0, acc[nt], false, false);
    }
  }
  // scores[m] += tanh(proj[m,n]) * ctx[n], reduced over all 512 n
#pragma unroll
  for (int nt = 0; nt < 4; ++nt) {
    const float cv = ctx[wave * 64 + nt * 16 + nl];
#pragma unroll
    for (int r = 0; r < 8; ++r)
      atomicAdd(&srow[mhi + r], tanhf(acc[nt][r]) * cv);
  }
  __syncthreads();
  if (tid < 16) {
    const int t = t0 + tid;
    scores[(size_t)b * Tn + t] = (t < len) ? srow[tid] : -INFINITY;
  }
}

__global__ __launch_bounds__(256) void softmax_kernel(
    const float* __restrict__ scores, float* __restrict__ attn)
{
  __shared__ float red[256];
  const int b = blockIdx.x, tid = threadIdx.x;
  const float* s = scores + (size_t)b * Tn;
  const float s0 = s[tid], s1 = s[tid + 256];
  red[tid] = fmaxf(s0, s1);
  __syncthreads();
  for (int off = 128; off > 0; off >>= 1) {
    if (tid < off) red[tid] = fmaxf(red[tid], red[tid + off]);
    __syncthreads();
  }
  const float mx = red[0];
  __syncthreads();
  const float e0 = __expf(s0 - mx), e1 = __expf(s1 - mx);
  red[tid] = e0 + e1;
  __syncthreads();
  for (int off = 128; off > 0; off >>= 1) {
    if (tid < off) red[tid] += red[tid + off];
    __syncthreads();
  }
  const float inv = 1.0f / red[0];
  attn[(size_t)b * Tn + tid]       = e0 * inv;
  attn[(size_t)b * Tn + tid + 256] = e1 * inv;
}

// sent = selu(sum_t attn*out); logits = sent @ Wl.T + bl
__global__ __launch_bounds__(256) void pool_kernel(
    const int* __restrict__ lengths, const float* __restrict__ attn,
    const _Float16* __restrict__ out_f, const _Float16* __restrict__ out_b,
    const float* __restrict__ Wl, const float* __restrict__ bl,
    float* __restrict__ logits)
{
  __shared__ float sent[2 * Hn];
  __shared__ float lg[NCn];
  const int b = blockIdx.x, tid = threadIdx.x;
  const int len = lengths[b];
  float a0 = 0.f, a1 = 0.f, a2 = 0.f, a3 = 0.f;
  for (int t = 0; t < Tn; ++t) {
    const float w = attn[(size_t)b * Tn + t];   // == 0 at masked positions
    const int trev = iclamp(len - 1 - t, 0, Tn - 1);
    const _Float16* rf = out_f + ((size_t)b * Tn + t) * Hn;
    const _Float16* rb = out_b + ((size_t)b * Tn + trev) * Hn;
    a0 += w * (float)rf[tid];
    a1 += w * (float)rf[tid + 256];
    a2 += w * (float)rb[tid];
    a3 += w * (float)rb[tid + 256];
  }
  sent[tid]       = seluf_(a0);
  sent[tid + 256] = seluf_(a1);
  sent[tid + 512] = seluf_(a2);
  sent[tid + 768] = seluf_(a3);
  if (tid < NCn) lg[tid] = bl[tid];
  __syncthreads();
  float p0 = 0.f, p1 = 0.f;
  for (int j = tid; j < 2 * Hn; j += 256) {
    const float sv = sent[j];
    p0 += sv * Wl[j];
    p1 += sv * Wl[2 * Hn + j];
  }
  atomicAdd(&lg[0], p0);
  atomicAdd(&lg[1], p1);
  __syncthreads();
  if (tid < NCn) logits[(size_t)b * NCn + tid] = lg[tid];
}

extern "C" void kernel_launch(void* const* d_in, const int* in_sizes, int n_in,
                              void* d_out, int out_size, void* d_ws, size_t ws_size,
                              hipStream_t stream) {
  const int*   tokens  = (const int*)d_in[0];
  const int*   lengths = (const int*)d_in[1];
  const float* emb     = (const float*)d_in[2];
  const float* Wih_f   = (const float*)d_in[3];
  const float* Whh_f   = (const float*)d_in[4];
  const float* bih_f   = (const float*)d_in[5];
  const float* bhh_f   = (const float*)d_in[6];
  const float* Wih_b   = (const float*)d_in[7];
  const float* Whh_b   = (const float*)d_in[8];
  const float* bih_b   = (const float*)d_in[9];
  const float* bhh_b   = (const float*)d_in[10];
  const float* Wp      = (const float*)d_in[11];
  const float* bp      = (const float*)d_in[12];
  const float* ctx     = (const float*)d_in[13];
  const float* Wl      = (const float*)d_in[14];
  const float* bl      = (const float*)d_in[15];

  // Workspace layout
  char* ws = (char*)d_ws;
  size_t off = 0;
  float* hbuf = (float*)(ws + off);           off += (size_t)4 * Bn * Hn * 4;       // 1 MB
  _Float16* out_f = (_Float16*)(ws + off);    off += (size_t)Bn * Tn * Hn * 2;      // 64 MB
  _Float16* out_b = (_Float16*)(ws + off);    off += (size_t)Bn * Tn * Hn * 2;      // 64 MB
  float* scores = (float*)(ws + off);         off += (size_t)Bn * Tn * 4;           // 256 KB
  _Float16* Wih16_f = (_Float16*)(ws + off);  off += (size_t)3 * Hn * En * 2;       // 1.5 MB
  _Float16* Whh16_f = (_Float16*)(ws + off);  off += (size_t)3 * Hn * Hn * 2;
  _Float16* Wih16_b = (_Float16*)(ws + off);  off += (size_t)3 * Hn * En * 2;
  _Float16* Whh16_b = (_Float16*)(ws + off);  off += (size_t)3 * Hn * Hn * 2;
  _Float16* Wp16 = (_Float16*)(ws + off);     off += (size_t)Cn * 2 * Hn * 2;       // 1 MB

  float* logits_out = (float*)d_out;                 // [B, NC]
  float* attn_out   = logits_out + Bn * NCn;         // [B, T]

  // 0) one-time (per launch) weight conversion to f16
  const int nW = 3 * Hn * En / 4, nP = Cn * 2 * Hn / 4;
  cvt_kernel<<<(nW + 255) / 256, 256, 0, stream>>>(Wih_f, Wih16_f, nW);
  cvt_kernel<<<(nW + 255) / 256, 256, 0, stream>>>(Whh_f, Whh16_f, nW);
  cvt_kernel<<<(nW + 255) / 256, 256, 0, stream>>>(Wih_b, Wih16_b, nW);
  cvt_kernel<<<(nW + 255) / 256, 256, 0, stream>>>(Whh_b, Whh16_b, nW);
  cvt_kernel<<<(nP + 255) / 256, 256, 0, stream>>>(Wp, Wp16, nP);

  // 1) zero hidden-state ping-pong buffers
  init_kernel<<<(4 * Bn * Hn + 255) / 256, 256, 0, stream>>>(hbuf, 4 * Bn * Hn);

  // 2) sequential bidirectional GRU scan (both directions per launch)
  for (int t = 0; t < Tn; ++t) {
    gru_step_kernel<<<dim3(Hn / 32, 2), 256, 0, stream>>>(
        t, tokens, lengths, emb,
        Wih16_f, Whh16_f, bih_f, bhh_f,
        Wih16_b, Whh16_b, bih_b, bhh_b,
        hbuf, out_f, out_b);
  }

  // 3) fused projection + tanh + context-dot -> masked scores
  proj_score_kernel<<<(Bn * Tn) / 16, 256, 0, stream>>>(
      lengths, out_f, out_b, Wp16, bp, ctx, scores);

  // 4) masked softmax over T -> attn (written directly into d_out)
  softmax_kernel<<<Bn, 256, 0, stream>>>(scores, attn_out);

  // 5) attention pooling + SELU + logits
  pool_kernel<<<Bn, 256, 0, stream>>>(lengths, attn_out, out_f, out_b, Wl, bl, logits_out);
}
